// CrossAttention_15522011808139
// MI455X (gfx1250) — matrix operand order repaired and checked by reading, not compiled, not run on hardware
//
#include <hip/hip_runtime.h>
#include <hip/hip_bf16.h>
#include <math.h>

// ---------------------------------------------------------------------------
// CDNA5 (gfx1250) wave32 WMMA cross-attention pipeline, v3.
// All GEMMs use v_wmma_f32_16x16x32_f16 (f16 in, f32 accum).
// v3: 32x64 wave tiles, fully-unrolled K loops (straight-line SSA so the
//     backend scheduler hoists fragment loads without register-rotation movs).
// ---------------------------------------------------------------------------

typedef __attribute__((ext_vector_type(16))) _Float16 v16h;
typedef __attribute__((ext_vector_type(8)))  float    v8f;
typedef __attribute__((ext_vector_type(4)))  float    v4f;

union Frag {
    v16h h;
    v4f  q[2];
};

#define WMMA_F16(a, b, c) \
    __builtin_amdgcn_wmma_f32_16x16x32_f16(false, (a), false, (b), (short)0, (c), false, false)

// --- fragment loaders -------------------------------------------------------
// A-matrix 16x32 f16 fragment (ISA 7.12.2): lane L holds row m=L%16;
// lanes 0-15: halfs 0..7 = K0..7, halfs 8..15 = K16..23
// lanes 16-31: halfs 0..7 = K8..15, halfs 8..15 = K24..31
// base points at element (row0, k0) of a row-major [M,K] matrix, ld in halfs.
__device__ __forceinline__ v16h ldA(const _Float16* base, int ld, int lane) {
    const int r  = lane & 15;
    const int kb = (lane & 16) >> 1;   // 0 or 8
    const _Float16* p = base + (size_t)r * ld + kb;
    Frag f;
    f.q[0] = *reinterpret_cast<const v4f*>(p);
    f.q[1] = *reinterpret_cast<const v4f*>(p + 16);
    return f.h;
}

// B-matrix 32x16 f16 fragment: lane L holds column n=L%16;
// lanes 0-15: halfs 0..7 = K0..7, halfs 8..15 = K16..23; lanes 16-31: K8..15, K24..31.
// Loaded from BT, a row-major [N,K] matrix; base points at (col0, k0).
__device__ __forceinline__ v16h ldB(const _Float16* bt, int ld, int lane) {
    const int n  = lane & 15;
    const int kb = (lane & 16) >> 1;   // 0 or 8
    const _Float16* p = bt + (size_t)n * ld + kb;
    Frag f;
    f.q[0] = *reinterpret_cast<const v4f*>(p);
    f.q[1] = *reinterpret_cast<const v4f*>(p + 16);
    return f.h;
}

// --- D-tile stores (C/D layout: VGPR r, lane L -> m = r + 8*(L/16), n = L%16)
__device__ __forceinline__ void storeD_f16_rowmajor(_Float16* D, int ld, int lane, const v8f d) {
    const int n = lane & 15, g = lane >> 4;
#pragma unroll
    for (int r = 0; r < 8; ++r)
        D[(size_t)(r + 8 * g) * ld + n] = (_Float16)d[r];
}

// store transposed: DT[n][m] row-major -> 8 contiguous halfs per lane (16B)
__device__ __forceinline__ void storeD_f16_trans(_Float16* DT, int ld, int lane, const v8f d) {
    const int n = lane & 15, g = lane >> 4;
    _Float16 tmp[8];
#pragma unroll
    for (int r = 0; r < 8; ++r) tmp[r] = (_Float16)d[r];
    *reinterpret_cast<v4f*>(DT + (size_t)n * ld + 8 * g) =
        *reinterpret_cast<const v4f*>(tmp);
}

// store transposed f32: DT[n][m] -> 8 contiguous floats per lane (2x16B)
__device__ __forceinline__ void storeD_f32_trans(float* DT, int ld, int lane, const v8f d) {
    const int n = lane & 15, g = lane >> 4;
    float tmp[8];
#pragma unroll
    for (int r = 0; r < 8; ++r) tmp[r] = d[r];
    v4f* dst = reinterpret_cast<v4f*>(DT + (size_t)n * ld + 8 * g);
    dst[0] = *reinterpret_cast<const v4f*>(tmp);
    dst[1] = *reinterpret_cast<const v4f*>(tmp + 4);
}

// --- core: one wave computes a 32(M) x 64(N) strip ---------------------------
// Fully unrolled over K: each fragment is a fresh SSA value, letting the
// backend scheduler hoist loads across WMMAs with no register rotation.
template <int K>
__device__ __forceinline__ void wave_gemm_32x64(const _Float16* __restrict__ A, int lda,
                                                const _Float16* __restrict__ BT, int ldb,
                                                int lane, v8f acc0[4], v8f acc1[4]) {
#pragma unroll
    for (int k0 = 0; k0 < K; k0 += 32) {
        v16h a0 = ldA(A + k0, lda, lane);
        v16h a1 = ldA(A + (size_t)16 * lda + k0, lda, lane);
#pragma unroll
        for (int t = 0; t < 4; ++t) {
            v16h b = ldB(BT + (size_t)(t * 16) * ldb + k0, ldb, lane);
            acc0[t] = WMMA_F16(a0, b, acc0[t]);
            acc1[t] = WMMA_F16(a1, b, acc1[t]);
        }
    }
}

// ===========================================================================
// Preprocessing kernels
// ===========================================================================

__global__ __launch_bounds__(256) void f32_to_f16_kernel(const float* __restrict__ src,
                                                         _Float16* __restrict__ dst, int n) {
    int i = blockIdx.x * 256 + threadIdx.x;
    if (i < n) dst[i] = (_Float16)src[i];
}

// rnorm[b][p] = sqrt(512) / max(||fmap[b,:,p]||2, eps)
__global__ __launch_bounds__(256) void fmap_rnorm_kernel(const float* __restrict__ fmap,
                                                         float* __restrict__ rn) {
    const int b = blockIdx.y;
    const int p = blockIdx.x * 256 + threadIdx.x;
    const float* base = fmap + (size_t)b * 512 * 1024 + p;
    float acc = 0.f;
#pragma unroll 8
    for (int c = 0; c < 512; ++c) {
        float v = base[(size_t)c * 1024];
        acc = fmaf(v, v, acc);
    }
    rn[b * 1024 + p] = 22.627416997969522f /* sqrt(512) */ / fmaxf(sqrtf(acc), 1e-12f);
}

// fnT[b][p][c] = fmap[b][c][p] * gamma[c] * rnorm[b][p]   (f16, K-contiguous in c)
__global__ __launch_bounds__(256) void fmap_to_fnT_kernel(const float* __restrict__ fmap,
                                                          const float* __restrict__ rn,
                                                          const float* __restrict__ gamma,
                                                          _Float16* __restrict__ fnT) {
    __shared__ float tile[32][33];
    const int b  = blockIdx.z;
    const int p0 = blockIdx.x * 32;
    const int c0 = blockIdx.y * 32;
    const int tx = threadIdx.x, ty = threadIdx.y;   // (32, 8)
#pragma unroll
    for (int i = 0; i < 4; ++i) {
        int c = c0 + ty + i * 8;
        tile[ty + i * 8][tx] = fmap[((size_t)b * 512 + c) * 1024 + p0 + tx] * gamma[c];
    }
    __syncthreads();
#pragma unroll
    for (int i = 0; i < 4; ++i) {
        int pr = ty + i * 8;
        int p  = p0 + pr;
        fnT[((size_t)b * 1024 + p) * 512 + c0 + tx] =
            (_Float16)(tile[tx][pr] * rn[b * 1024 + p]);
    }
}

// cn[b][n][c] = context * sqrt(768)/||row|| * gamma_ctx[c]  (f16)
__global__ __launch_bounds__(256) void ctx_norm_kernel(const float* __restrict__ ctx,
                                                       const float* __restrict__ gamma,
                                                       _Float16* __restrict__ cnH) {
    const int row = blockIdx.x;                 // b*256 + n
    const float* base = ctx + (size_t)row * 768;
    float acc = 0.f;
    for (int c = threadIdx.x; c < 768; c += 256) {
        float v = base[c];
        acc = fmaf(v, v, acc);
    }
    __shared__ float red[256];
    red[threadIdx.x] = acc;
    __syncthreads();
    for (int s = 128; s > 0; s >>= 1) {
        if (threadIdx.x < s) red[threadIdx.x] += red[threadIdx.x + s];
        __syncthreads();
    }
    const float rs = 27.712812921102035f /* sqrt(768) */ / fmaxf(sqrtf(red[0]), 1e-12f);
    for (int c = threadIdx.x; c < 768; c += 256)
        cnH[(size_t)row * 768 + c] = (_Float16)(base[c] * rs * gamma[c]);
}

// ===========================================================================
// GEMM kernels (4 waves / block, each wave a 32x64 strip => block = 128x64)
// ===========================================================================

// qT[b][p][o] = sum_c fnT[b][p][c] * Wq[o][c]       M=1024 N=512 K=512
__global__ __launch_bounds__(128) void gemm_q_kernel(const _Float16* __restrict__ fnT,
                                                     const _Float16* __restrict__ WqH,
                                                     _Float16* __restrict__ qT) {
    const int b = blockIdx.x, mt = blockIdx.y, nt = blockIdx.z;
    const int wave = threadIdx.x >> 5, lane = threadIdx.x & 31;
    const int row0 = mt * 128 + wave * 32;  // p
    const int col0 = nt * 64;               // o
    v8f acc0[4] = {}, acc1[4] = {};
    wave_gemm_32x64<512>(fnT + ((size_t)b * 1024 + row0) * 512, 512,
                         WqH + (size_t)col0 * 512, 512, lane, acc0, acc1);
#pragma unroll
    for (int t = 0; t < 4; ++t) {
        storeD_f16_rowmajor(qT + ((size_t)b * 1024 + row0) * 512 + col0 + t * 16,
                            512, lane, acc0[t]);
        storeD_f16_rowmajor(qT + ((size_t)b * 1024 + row0 + 16) * 512 + col0 + t * 16,
                            512, lane, acc1[t]);
    }
}

// kv[b][n][o] = sum_c cn[b][n][c] * Wkv[o][c]       M=256 N=1024 K=768
// o <  512 -> Kmat[b][n][o]        (row-major, A-side of score GEMM)
// o >= 512 -> VT[b][o-512][n]      (transposed, B-side of PV GEMM)
__global__ __launch_bounds__(128) void gemm_kv_kernel(const _Float16* __restrict__ cnH,
                                                      const _Float16* __restrict__ WkvH,
                                                      _Float16* __restrict__ Kmat,
                                                      _Float16* __restrict__ VT) {
    const int b = blockIdx.x, mt = blockIdx.y, nt = blockIdx.z;
    const int wave = threadIdx.x >> 5, lane = threadIdx.x & 31;
    const int row0 = mt * 128 + wave * 32;  // n_ctx
    const int col0 = nt * 64;               // o
    v8f acc0[4] = {}, acc1[4] = {};
    wave_gemm_32x64<768>(cnH + ((size_t)b * 256 + row0) * 768, 768,
                         WkvH + (size_t)col0 * 768, 768, lane, acc0, acc1);
#pragma unroll
    for (int t = 0; t < 4; ++t) {
        const int o = col0 + t * 16;
        if (o < 512) {
            storeD_f16_rowmajor(Kmat + ((size_t)b * 256 + row0) * 512 + o,
                                512, lane, acc0[t]);
            storeD_f16_rowmajor(Kmat + ((size_t)b * 256 + row0 + 16) * 512 + o,
                                512, lane, acc1[t]);
        } else {
            storeD_f16_trans(VT + ((size_t)b * 512 + (o - 512)) * 256 + row0,
                             256, lane, acc0[t]);
            storeD_f16_trans(VT + ((size_t)b * 512 + (o - 512)) * 256 + row0 + 16,
                             256, lane, acc1[t]);
        }
    }
}

// ===========================================================================
// Attention: per (b, h), each wave owns 16 query rows x all 256 keys.
// Orientation: simT = K x Q^T  => lane index = query column i, so softmax
// reductions are VALU + one shfl_xor(16), and P^T register tiles are already
// the A-fragment layout for the P x V GEMM (no LDS transpose).
// ===========================================================================
__global__ __launch_bounds__(128) void attn_kernel(const _Float16* __restrict__ qT,
                                                   const _Float16* __restrict__ Kmat,
                                                   const _Float16* __restrict__ VT,
                                                   const unsigned char* __restrict__ mask,
                                                   _Float16* __restrict__ attnT) {
    const int b = blockIdx.x, h = blockIdx.y, it = blockIdx.z;
    const int wave = threadIdx.x >> 5, lane = threadIdx.x & 31;
    const int i0 = it * 64 + wave * 16;       // query row base
    const int g  = lane >> 4;

    const _Float16* qb = qT   + (size_t)b * 1024 * 512;
    const _Float16* kb = Kmat + (size_t)b * 256 * 512;
    const _Float16* vb = VT   + (size_t)b * 512 * 256;

    // B-side fragments: q rows i0..i0+15, head slice, two 32-wide K chunks (d=64)
    const v16h qf0 = ldB(qb + (size_t)i0 * 512 + h * 64 + 0,  512, lane);
    const v16h qf1 = ldB(qb + (size_t)i0 * 512 + h * 64 + 32, 512, lane);

    // scores: 16 j-tiles x (16j x 16i), K=64 -> 32 WMMAs
    v8f S[16];
#pragma unroll
    for (int jt = 0; jt < 16; ++jt) {
        v16h a0 = ldA(kb + (size_t)(jt * 16) * 512 + h * 64 + 0,  512, lane);
        v16h a1 = ldA(kb + (size_t)(jt * 16) * 512 + h * 64 + 32, 512, lane);
        v8f c = {};
        c = WMMA_F16(a0, qf0, c);
        c = WMMA_F16(a1, qf1, c);
        S[jt] = c;
    }

    // scale + mask + softmax over j (tile row m = r + 8g => j = jt*16 + r + 8g)
    const float scale = 0.125f;               // 64^-0.5
    const float NEG = -3.0e38f;
    const unsigned char* mrow = mask + b * 256;
    float mx = NEG;
#pragma unroll
    for (int jt = 0; jt < 16; ++jt)
#pragma unroll
        for (int r = 0; r < 8; ++r) {
            const int j = jt * 16 + r + 8 * g;
            float s = S[jt][r] * scale;
            s = mrow[j] ? s : NEG;
            S[jt][r] = s;
            mx = fmaxf(mx, s);
        }
    mx = fmaxf(mx, __shfl_xor(mx, 16, 32));
    float sum = 0.f;
#pragma unroll
    for (int jt = 0; jt < 16; ++jt)
#pragma unroll
        for (int r = 0; r < 8; ++r) {
            float e = __expf(S[jt][r] - mx);
            S[jt][r] = e;
            sum += e;
        }
    sum += __shfl_xor(sum, 16, 32);
    const float inv = 1.0f / sum;

    // P^T tiles -> A fragments: chunk jc covers j = 32jc..32jc+31 = tiles 2jc, 2jc+1
    // lane holds j = 32jc + 8g + r (halfs 0..7) and +16 (halfs 8..15): exact A layout.
    v16h P[8];
#pragma unroll
    for (int jc = 0; jc < 8; ++jc) {
        Frag f;
#pragma unroll
        for (int r = 0; r < 8; ++r) {
            f.h[r]     = (_Float16)(S[2 * jc][r]     * inv);
            f.h[r + 8] = (_Float16)(S[2 * jc + 1][r] * inv);
        }
        P[jc] = f.h;
    }

    // O[i, d] = P x V : 4 d-tile accumulators, K=256 -> 32 WMMAs (jc outer so
    // each register-resident P fragment feeds 4 WMMAs).
    v8f o[4] = {};
#pragma unroll
    for (int jc = 0; jc < 8; ++jc) {
#pragma unroll
        for (int dt = 0; dt < 4; ++dt) {
            v16h bf = ldB(vb + (size_t)(h * 64 + dt * 16) * 256 + jc * 32, 256, lane);
            o[dt] = WMMA_F16(P[jc], bf, o[dt]);
        }
    }
#pragma unroll
    for (int dt = 0; dt < 4; ++dt)
        storeD_f16_rowmajor(attnT + ((size_t)b * 1024 + i0) * 512 + h * 64 + dt * 16,
                            512, lane, o[dt]);
}

// out[b][o][p] = sum_i attnT[b][p][i] * Wout[o][i]   M=1024(p) N=512(o) K=512
// D-tile (m=p, n=o) stored transposed -> 8 contiguous f32 per lane.
__global__ __launch_bounds__(128) void gemm_out_kernel(const _Float16* __restrict__ attnT,
                                                       const _Float16* __restrict__ WoutH,
                                                       float* __restrict__ out) {
    const int b = blockIdx.x, mt = blockIdx.y, nt = blockIdx.z;
    const int wave = threadIdx.x >> 5, lane = threadIdx.x & 31;
    const int row0 = mt * 128 + wave * 32;  // p
    const int col0 = nt * 64;               // o
    v8f acc0[4] = {}, acc1[4] = {};
    wave_gemm_32x64<512>(attnT + ((size_t)b * 1024 + row0) * 512, 512,
                         WoutH + (size_t)col0 * 512, 512, lane, acc0, acc1);
#pragma unroll
    for (int t = 0; t < 4; ++t) {
        storeD_f32_trans(out + ((size_t)b * 512 + col0 + t * 16) * 1024 + row0,
                         1024, lane, acc0[t]);
        storeD_f32_trans(out + ((size_t)b * 512 + col0 + t * 16) * 1024 + row0 + 16,
                         1024, lane, acc1[t]);
    }
}

// ===========================================================================
// Host launch
// ===========================================================================
extern "C" void kernel_launch(void* const* d_in, const int* in_sizes, int n_in,
                              void* d_out, int out_size, void* d_ws, size_t ws_size,
                              hipStream_t stream) {
    const float* fmap     = (const float*)d_in[0];   // (32, 512, 32, 32)
    const float* context  = (const float*)d_in[1];   // (32, 256, 768)
    const unsigned char* mask = (const unsigned char*)d_in[2]; // (32, 256) bool
    const float* gamma_f  = (const float*)d_in[3];   // (512,)
    const float* gamma_c  = (const float*)d_in[4];   // (768,)
    const float* Wq       = (const float*)d_in[5];   // (512, 512)
    const float* Wkv      = (const float*)d_in[6];   // (1024, 768)
    const float* Wout     = (const float*)d_in[7];   // (512, 512)
    float* out            = (float*)d_out;           // (32, 512, 32, 32)

    // workspace layout (256B aligned)
    char* ws = (char*)d_ws;
    size_t off = 0;
    auto take = [&](size_t bytes) {
        char* p = ws + off;
        off += (bytes + 255) & ~size_t(255);
        return p;
    };
    _Float16* WqH   = (_Float16*)take((size_t)512 * 512 * 2);
    _Float16* WkvH  = (_Float16*)take((size_t)1024 * 768 * 2);
    _Float16* WoutH = (_Float16*)take((size_t)512 * 512 * 2);
    float*    rnorm = (float*)   take((size_t)32 * 1024 * 4);
    _Float16* fnT   = (_Float16*)take((size_t)32 * 1024 * 512 * 2);
    _Float16* cnH   = (_Float16*)take((size_t)32 * 256 * 768 * 2);
    _Float16* qT    = (_Float16*)take((size_t)32 * 1024 * 512 * 2);
    _Float16* Kmat  = (_Float16*)take((size_t)32 * 256 * 512 * 2);
    _Float16* VT    = (_Float16*)take((size_t)32 * 512 * 256 * 2);
    _Float16* attnT = (_Float16*)take((size_t)32 * 1024 * 512 * 2);
    (void)ws_size; (void)n_in; (void)in_sizes; (void)out_size;

    // weights -> f16
    f32_to_f16_kernel<<<(512 * 512 + 255) / 256, 256, 0, stream>>>(Wq, WqH, 512 * 512);
    f32_to_f16_kernel<<<(1024 * 768 + 255) / 256, 256, 0, stream>>>(Wkv, WkvH, 1024 * 768);
    f32_to_f16_kernel<<<(512 * 512 + 255) / 256, 256, 0, stream>>>(Wout, WoutH, 512 * 512);

    // normalizations
    fmap_rnorm_kernel<<<dim3(4, 32), 256, 0, stream>>>(fmap, rnorm);
    fmap_to_fnT_kernel<<<dim3(32, 16, 32), dim3(32, 8), 0, stream>>>(fmap, rnorm, gamma_f, fnT);
    ctx_norm_kernel<<<32 * 256, 256, 0, stream>>>(context, gamma_c, cnH);

    // projections (block = 128 rows x 64 cols, 4 waves of 32x64)
    gemm_q_kernel <<<dim3(32, 8, 8),  128, 0, stream>>>(fnT, WqH, qT);
    gemm_kv_kernel<<<dim3(32, 2, 16), 128, 0, stream>>>(cnH, WkvH, Kmat, VT);

    // attention
    attn_kernel<<<dim3(32, 8, 16), 128, 0, stream>>>(qT, Kmat, VT, mask, attnT);

    // output projection
    gemm_out_kernel<<<dim3(32, 8, 8), 128, 0, stream>>>(attnT, WoutH, out);
}
